// MultiHeadAttention_35253091566074
// MI455X (gfx1250) — matrix operand order repaired and checked by reading, not compiled, not run on hardware
//
#include <hip/hip_runtime.h>
#include <stdint.h>

typedef __attribute__((ext_vector_type(16))) _Float16 v16h;
typedef __attribute__((ext_vector_type(8)))  float    v8f;

#define D_MODEL 1024
#define N_HEADS 16
#define DH      64
#define BATCH   2
#define T_SEQ   2048
#define M_TOT   (BATCH * T_SEQ)   // 4096

// ---------------------------------------------------------------------------
// WMMA fragment loaders per CDNA5 ISA 7.12.2 lane layouts (wave32).
// A (16x32 f16): lane L holds row M = L%16.
//   lanes 0-15 : elems 0..7 -> K=k0+0..7,  elems 8..15 -> K=k0+16..23
//   lanes 16-31: elems 0..7 -> K=k0+8..15, elems 8..15 -> K=k0+24..31
// B (32x16 f16): lane L holds col N = L%16; same K mapping as A
// C/D (16x16 f32, 8 VGPRs): vgpr v -> row M = v + (L<16?0:8), col N = L%16
// ---------------------------------------------------------------------------

__device__ __forceinline__ v16h load_a_frag_f32(const float* base, int ld,
                                                int row, int k0, int lane) {
  const float* r = base + (size_t)row * ld;
  const int lo = k0 + ((lane & 16) ? 8 : 0);
  const int hi = k0 + ((lane & 16) ? 24 : 16);
  v16h a;
#pragma unroll
  for (int i = 0; i < 8; ++i) a[i] = (_Float16)r[lo + i];
#pragma unroll
  for (int i = 0; i < 8; ++i) a[8 + i] = (_Float16)r[hi + i];
  return a;
}

__device__ __forceinline__ v16h load_a_frag_f16(const _Float16* base, int ld,
                                                int row, int k0, int lane) {
  const _Float16* r = base + (size_t)row * ld;
  const int lo = k0 + ((lane & 16) ? 8 : 0);
  const int hi = k0 + ((lane & 16) ? 24 : 16);
  v16h a;
#pragma unroll
  for (int i = 0; i < 8; ++i) a[i] = r[lo + i];
#pragma unroll
  for (int i = 0; i < 8; ++i) a[8 + i] = r[hi + i];
  return a;
}

// B fragment where K runs contiguously in memory starting at row_n[k0].
__device__ __forceinline__ v16h load_b_frag_rowk_f32(const float* row_n,
                                                     int k0, int lane) {
  const int lo = k0 + ((lane & 16) ? 8 : 0);
  const int hi = k0 + ((lane & 16) ? 24 : 16);
  v16h b;
#pragma unroll
  for (int i = 0; i < 8; ++i) b[i] = (_Float16)row_n[lo + i];
#pragma unroll
  for (int i = 0; i < 8; ++i) b[8 + i] = (_Float16)row_n[hi + i];
  return b;
}

__device__ __forceinline__ v16h load_b_frag_rowk_f16(const _Float16* row_n,
                                                     int k0, int lane) {
  const int lo = k0 + ((lane & 16) ? 8 : 0);
  const int hi = k0 + ((lane & 16) ? 24 : 16);
  v16h b;
#pragma unroll
  for (int i = 0; i < 8; ++i) b[i] = row_n[lo + i];
#pragma unroll
  for (int i = 0; i < 8; ++i) b[8 + i] = row_n[hi + i];
  return b;
}

__device__ __forceinline__ v8f wmma_f16(v16h a, v16h b, v8f c) {
  return __builtin_amdgcn_wmma_f32_16x16x32_f16(false, a, false, b,
                                                (short)0, c, false, false);
}

// Async global->LDS copy (16B per lane), CDNA5 GLOBAL_LOAD_ASYNC_TO_LDS_B128.
// Tracked with ASYNCcnt; wait with s_wait_asynccnt before consuming.
__device__ __forceinline__ void async_copy_b128(uint32_t lds_byte_off,
                                                const void* gptr) {
  asm volatile("global_load_async_to_lds_b128 %0, %1, off"
               :
               : "v"(lds_byte_off), "v"(gptr)
               : "memory");
}
__device__ __forceinline__ void wait_asynccnt0() {
  asm volatile("s_wait_asynccnt 0x0" ::: "memory");
}

// ---------------------------------------------------------------------------
// Kernels 1/3: Y = X @ W^T.  X:[M,K], W:[N,K] f32.
//   X_IS_H: 0 = X fp32, 1 = X f16
//   OUT_MODE: 0 = f16 into [B,H,T,DH] (Q,K)
//             1 = f32 flat [M,N]       (final out)
//             2 = f16 into [B,H,DH,T]  (V transposed for P@V B-fragments)
// 2x2 register blocking: each wave owns a 32x32 tile (4 WMMA accumulators,
// 4 WMMAs per k-step from 2 A-frags + 2 B-frags -> 2x arithmetic intensity
// and 4 independent XDL chains). Block = 8 waves (2x4) = 64x128 tile.
// Grid: (M/64, N/128).
// ---------------------------------------------------------------------------
template <int X_IS_H, int OUT_MODE>
__device__ __forceinline__ void store_tile(v8f c, _Float16* Yh, float* Yf,
                                           int m0, int n0, int lane) {
  const int n = n0 + (lane & 15);
#pragma unroll
  for (int v = 0; v < 8; ++v) {
    const int m = m0 + v + ((lane & 16) ? 8 : 0);
    if (OUT_MODE == 0) {
      const int bb = m / T_SEQ, t = m % T_SEQ;
      const int h = n >> 6, d = n & 63;
      Yh[(((size_t)bb * N_HEADS + h) * T_SEQ + t) * DH + d] = (_Float16)c[v];
    } else if (OUT_MODE == 2) {
      const int bb = m / T_SEQ, t = m % T_SEQ;
      const int h = n >> 6, d = n & 63;
      Yh[(((size_t)bb * N_HEADS + h) * DH + d) * T_SEQ + t] = (_Float16)c[v];
    } else {
      Yf[(size_t)m * D_MODEL + n] = c[v];
    }
  }
}

template <int X_IS_H, int OUT_MODE>
__global__ __launch_bounds__(256) void gemm_xWt_kernel(
    const float* __restrict__ Xf, const _Float16* __restrict__ Xh,
    const float* __restrict__ W, _Float16* __restrict__ Yh,
    float* __restrict__ Yf) {
  const int lane = threadIdx.x & 31;
  const int wave = threadIdx.x >> 5;
  const int mi = wave >> 2;  // 0..1 -> 32-row group
  const int ni = wave & 3;   // 0..3 -> 32-col group
  const int m0 = blockIdx.x * 64 + mi * 32;
  const int n0 = blockIdx.y * 128 + ni * 32;
  const int ar0 = m0 + (lane & 15);
  const int ar1 = m0 + 16 + (lane & 15);
  const float* wrow0 = W + (size_t)(n0 + (lane & 15)) * D_MODEL;
  const float* wrow1 = W + (size_t)(n0 + 16 + (lane & 15)) * D_MODEL;

  v8f c00 = {}, c01 = {}, c10 = {}, c11 = {};
#pragma unroll 2
  for (int k0 = 0; k0 < D_MODEL; k0 += 32) {
    if (k0 + 32 < D_MODEL) {
      __builtin_prefetch(wrow0 + k0 + 32, 0, 3);
      __builtin_prefetch(wrow1 + k0 + 32, 0, 3);
    }
    v16h a0 = X_IS_H ? load_a_frag_f16(Xh, D_MODEL, ar0, k0, lane)
                     : load_a_frag_f32(Xf, D_MODEL, ar0, k0, lane);
    v16h a1 = X_IS_H ? load_a_frag_f16(Xh, D_MODEL, ar1, k0, lane)
                     : load_a_frag_f32(Xf, D_MODEL, ar1, k0, lane);
    v16h b0 = load_b_frag_rowk_f32(wrow0, k0, lane);
    v16h b1 = load_b_frag_rowk_f32(wrow1, k0, lane);
    c00 = wmma_f16(a0, b0, c00);
    c01 = wmma_f16(a0, b1, c01);
    c10 = wmma_f16(a1, b0, c10);
    c11 = wmma_f16(a1, b1, c11);
  }

  store_tile<X_IS_H, OUT_MODE>(c00, Yh, Yf, m0,      n0,      lane);
  store_tile<X_IS_H, OUT_MODE>(c01, Yh, Yf, m0,      n0 + 16, lane);
  store_tile<X_IS_H, OUT_MODE>(c10, Yh, Yf, m0 + 16, n0,      lane);
  store_tile<X_IS_H, OUT_MODE>(c11, Yh, Yf, m0 + 16, n0 + 16, lane);
}

// ---------------------------------------------------------------------------
// Kernel 2: fused attention for one (b, h, 32-query-row) block.
// 256 KB dynamic LDS score panel + 4 KB async-staged Q tile.
// ---------------------------------------------------------------------------
__global__ __launch_bounds__(256) void attn_kernel(
    const _Float16* __restrict__ Qw, const _Float16* __restrict__ Kw,
    const _Float16* __restrict__ Vt, float* __restrict__ attn_out,
    _Float16* __restrict__ ctx) {
  extern __shared__ char smem[];
  float* S = (float*)smem;                                // [32][T_SEQ]
  _Float16* Qs = (_Float16*)(smem + 32 * T_SEQ * 4);      // [32][DH]

  const int lane = threadIdx.x & 31;
  const int wave = threadIdx.x >> 5;

  const int qb = blockIdx.x % (T_SEQ / 32);
  const int bh = blockIdx.x / (T_SEQ / 32);
  const int h = bh % N_HEADS;
  const int b = bh / N_HEADS;
  const int q0 = qb * 32;

  const _Float16* Qh = Qw + ((size_t)b * N_HEADS + h) * T_SEQ * DH;
  const _Float16* Kh = Kw + ((size_t)b * N_HEADS + h) * T_SEQ * DH;
  const _Float16* Vh = Vt + ((size_t)b * N_HEADS + h) * DH * T_SEQ;  // [DH][T]

  // ---- Stage Q tile (32 x 64 f16 = 4 KB) into LDS via async DMA.
  {
    const int row = threadIdx.x >> 3;          // 0..31
    const int seg = threadIdx.x & 7;           // 0..7, 8 halves each
    const _Float16* src = Qh + (size_t)(q0 + row) * DH + seg * 8;
    const uint32_t dst = (uint32_t)(uintptr_t)(Qs + row * DH + seg * 8);
    async_copy_b128(dst, src);
    wait_asynccnt0();
  }
  __syncthreads();

  // ---- Phase 1: S = (Q K^T) * 1/sqrt(DH). Wave owns cols [wave*256, +256).
  v16h qa[2][2];
#pragma unroll
  for (int mi = 0; mi < 2; ++mi)
#pragma unroll
    for (int ks = 0; ks < 2; ++ks)
      qa[mi][ks] = load_a_frag_f16(Qs, DH, mi * 16 + (lane & 15), ks * 32, lane);

  for (int jt = 0; jt < 16; ++jt) {
    const int j0 = wave * 256 + jt * 16;
    const _Float16* krow = Kh + (size_t)(j0 + (lane & 15)) * DH;
    v16h kb0 = load_b_frag_rowk_f16(krow, 0, lane);
    v16h kb1 = load_b_frag_rowk_f16(krow, 32, lane);
#pragma unroll
    for (int mi = 0; mi < 2; ++mi) {
      v8f c = {};
      c = wmma_f16(qa[mi][0], kb0, c);
      c = wmma_f16(qa[mi][1], kb1, c);
      const int col = j0 + (lane & 15);
#pragma unroll
      for (int v = 0; v < 8; ++v) {
        const int row = mi * 16 + v + ((lane & 16) ? 8 : 0);
        S[(size_t)row * T_SEQ + col] = c[v] * 0.125f;  // 1/sqrt(64)
      }
    }
  }
  __syncthreads();

  // ---- Phase 2: softmax over each of 32 rows (4 rows per wave), write attn.
  for (int r = 0; r < 4; ++r) {
    const int row = wave * 4 + r;
    float* srow = S + (size_t)row * T_SEQ;
    float mx = -3.402823e38f;
    for (int j = lane; j < T_SEQ; j += 32) mx = fmaxf(mx, srow[j]);
#pragma unroll
    for (int off = 16; off > 0; off >>= 1)
      mx = fmaxf(mx, __shfl_xor(mx, off, 32));
    float sum = 0.0f;
    for (int j = lane; j < T_SEQ; j += 32) {
      float e = __expf(srow[j] - mx);
      srow[j] = e;
      sum += e;
    }
#pragma unroll
    for (int off = 16; off > 0; off >>= 1) sum += __shfl_xor(sum, off, 32);
    const float inv = 1.0f / sum;
    float* arow =
        attn_out + ((((size_t)b * N_HEADS + h) * T_SEQ) + q0 + row) * T_SEQ;
    for (int j = lane; j < T_SEQ; j += 32) {
      float p = srow[j] * inv;
      srow[j] = p;       // keep normalized P in LDS for phase 3
      arow[j] = p;       // emit attention matrix (fp32 output #2)
    }
  }
  __syncthreads();

  // ---- Phase 3: O = P @ V. Wave (mi,ni) owns a 16x16 tile of 32x64 output.
  // V is stored transposed [DH][T], so B-fragment loads are contiguous in K.
  const int mi = wave >> 2;
  const int ni = wave & 3;
  const _Float16* vrow = Vh + (size_t)(ni * 16 + (lane & 15)) * T_SEQ;
  v8f c = {};
  for (int k0 = 0; k0 < T_SEQ; k0 += 32) {
    // A fragment from LDS (f32 -> f16 on the fly)
    const float* prow = S + (size_t)(mi * 16 + (lane & 15)) * T_SEQ;
    const int lo = k0 + ((lane & 16) ? 8 : 0);
    const int hi = k0 + ((lane & 16) ? 24 : 16);
    v16h a;
#pragma unroll
    for (int i = 0; i < 8; ++i) a[i] = (_Float16)prow[lo + i];
#pragma unroll
    for (int i = 0; i < 8; ++i) a[8 + i] = (_Float16)prow[hi + i];
    v16h bf = load_b_frag_rowk_f16(vrow, k0, lane);
    c = wmma_f16(a, bf, c);
  }
  const int d = ni * 16 + (lane & 15);
#pragma unroll
  for (int v = 0; v < 8; ++v) {
    const int row = mi * 16 + v + ((lane & 16) ? 8 : 0);
    const int t = q0 + row;
    // ctx stored as [B*T, D_MODEL] f16, cols grouped by head -> ready for W_o
    ctx[((size_t)b * T_SEQ + t) * D_MODEL + h * DH + d] = (_Float16)c[v];
  }
}

// ---------------------------------------------------------------------------
extern "C" void kernel_launch(void* const* d_in, const int* in_sizes, int n_in,
                              void* d_out, int out_size, void* d_ws,
                              size_t ws_size, hipStream_t stream) {
  (void)in_sizes; (void)n_in; (void)out_size; (void)ws_size;
  const float* x_q  = (const float*)d_in[0];
  const float* x_kv = (const float*)d_in[1];
  const float* w_q  = (const float*)d_in[2];
  const float* w_k  = (const float*)d_in[3];
  const float* w_v  = (const float*)d_in[4];
  const float* w_o  = (const float*)d_in[5];

  float* out  = (float*)d_out;                         // [2,2048,1024]
  float* attn = out + (size_t)M_TOT * D_MODEL;         // [2,16,2048,2048]

  _Float16* Qw  = (_Float16*)d_ws;                     // 8 MB each
  _Float16* Kw  = Qw + (size_t)M_TOT * D_MODEL;
  _Float16* Vw  = Kw + (size_t)M_TOT * D_MODEL;        // [B,H,DH,T]
  _Float16* Ctx = Vw + (size_t)M_TOT * D_MODEL;

  dim3 gridG(M_TOT / 64, D_MODEL / 128);
  dim3 blk(256);

  // QKV projections: fp32 x @ W^T -> f16 (Q,K: [B,H,T,DH]; V: [B,H,DH,T])
  gemm_xWt_kernel<0, 0><<<gridG, blk, 0, stream>>>(x_q, nullptr, w_q, Qw, nullptr);
  gemm_xWt_kernel<0, 0><<<gridG, blk, 0, stream>>>(x_kv, nullptr, w_k, Kw, nullptr);
  gemm_xWt_kernel<0, 2><<<gridG, blk, 0, stream>>>(x_kv, nullptr, w_v, Vw, nullptr);

  // Fused attention: one block per (b, h, 32 q-rows); 260 KB dynamic LDS.
  const size_t lds_bytes = (size_t)32 * T_SEQ * 4 + (size_t)32 * DH * 2;
  attn_kernel<<<BATCH * N_HEADS * (T_SEQ / 32), blk, lds_bytes, stream>>>(
      Qw, Kw, Vw, attn, Ctx);

  // Output projection: f16 ctx @ W_o^T -> fp32 out
  gemm_xWt_kernel<1, 1><<<gridG, blk, 0, stream>>>(nullptr, Ctx, w_o, nullptr, out);
}
